// CustomBiGRU_62027917689185
// MI455X (gfx1250) — matrix operand order repaired and checked by reading, not compiled, not run on hardware
//
#include <hip/hip_runtime.h>
#include <cstdint>

typedef __attribute__((ext_vector_type(16))) __bf16 v16bf;
typedef __attribute__((ext_vector_type(8)))  float  v8f;
typedef __bf16 bf16_t;
typedef int v4i_ls __attribute__((vector_size(16)));  // matches async builtin pointee

#define T_SEQ  256
#define BATCH  64
#define H_DIM  1024
#define IN_DIM 1024
#define KC     32     // K-chunk per WMMA (bf16 16x16x32)
#define LDS_LD 40     // padded LDS row stride in halfs (80 bytes, 16B aligned)
#define JB     32     // hidden columns per gru_step block

#if __has_builtin(__builtin_amdgcn_global_load_async_to_lds_b128)
#define HAVE_ASYNC_LDS 1
#else
#define HAVE_ASYNC_LDS 0
#endif

// ---------------------------------------------------------------------------
// 16-byte global->LDS stage: async DMA when available (ASYNCcnt path),
// otherwise caller uses register staging (see kernels).
// ---------------------------------------------------------------------------
__device__ __forceinline__ void async_cp16(const bf16_t* g, __bf16* l) {
#if HAVE_ASYNC_LDS
  __builtin_amdgcn_global_load_async_to_lds_b128(
      (__attribute__((address_space(1))) v4i_ls*)(unsigned long long)g,
      (__attribute__((address_space(3))) v4i_ls*)(unsigned long long)l, 0, 0);
#else
  *(uint4*)l = *(const uint4*)g;
#endif
}

__device__ __forceinline__ void stage_sync() {
#if HAVE_ASYNC_LDS
#if __has_builtin(__builtin_amdgcn_s_wait_asynccnt)
  __builtin_amdgcn_s_wait_asynccnt(0);
#else
  asm volatile("s_wait_asynccnt 0x0" ::: "memory");
#endif
#endif
  __syncthreads();
}

// ---------------------------------------------------------------------------
// Fragment loaders per CDNA5 ISA 7.12.2 (wave32 bf16 16x16x32 layouts).
// A (16x32, MxK): lane l -> m=l%16, khi=l/16; halves = src[m][khi*8 + 0..7]
//                 and src[m][16 + khi*8 + 0..7].
// B (32x16, KxN) built from U stored (N,K) row-major (B[k][n] = U[n][k]):
//                 lane l -> n=l%16, khi=l/16; halves = U[n][khi*8 + 0..7]
//                 and U[n][16 + khi*8 + 0..7].
// ---------------------------------------------------------------------------
__device__ __forceinline__ v16bf load_a_frag(const __bf16* base, int ld) {
  int lane = threadIdx.x & 31;
  int m = lane & 15, khi = lane >> 4;
  const __bf16* p0 = base + m * ld + khi * 8;
  const __bf16* p1 = base + m * ld + 16 + khi * 8;
  v16bf a;
#pragma unroll
  for (int i = 0; i < 8; ++i) a[i] = p0[i];
#pragma unroll
  for (int i = 0; i < 8; ++i) a[8 + i] = p1[i];
  return a;
}

__device__ __forceinline__ v16bf load_b_frag(const __bf16* base, int ld) {
  int lane = threadIdx.x & 31;
  int n = lane & 15, khi = lane >> 4;
  const __bf16* p0 = base + n * ld + khi * 8;
  const __bf16* p1 = base + n * ld + 16 + khi * 8;
  v16bf b;
#pragma unroll
  for (int i = 0; i < 8; ++i) b[i] = p0[i];
#pragma unroll
  for (int i = 0; i < 8; ++i) b[8 + i] = p1[i];
  return b;
}

__device__ __forceinline__ v8f wmma_bf16(v16bf a, v16bf b, v8f c) {
  return __builtin_amdgcn_wmma_f32_16x16x32_bf16(false, a, false, b,
                                                 (short)0, c, false, false);
}

// ---------------------------------------------------------------------------
// Big GEMM with bias:  C(M,N) = A(M,K) * Bm(N,K)^T + bias(N)     (f32 out)
// 128x128 tile / 256-thread block; double-buffered LDS software pipeline.
// ---------------------------------------------------------------------------
__global__ __launch_bounds__(256) void gemm_bf16_bias(
    const bf16_t* __restrict__ A, const bf16_t* __restrict__ Bm,
    const float* __restrict__ bias, float* __restrict__ C,
    int M, int N, int K)
{
  __shared__ __align__(16) __bf16 sA[2][128][LDS_LD];
  __shared__ __align__(16) __bf16 sB[2][128][LDS_LD];

  const int bn = blockIdx.x, bm = blockIdx.y;
  const int tid = threadIdx.x;
  const int w = tid >> 5, lane = tid & 31;

  // each thread stages two 16B vectors per matrix per chunk
  const int r0 = tid >> 2;            // row 0..63 (second row = r0+64)
  const int v8off = (tid & 3) * 8;    // column offset in halfs
  const bf16_t* Ab  = A  + (size_t)(bm * 128 + r0) * K + v8off;
  const bf16_t* Ab2 = Ab + (size_t)64 * K;
  const bf16_t* Bb  = Bm + (size_t)(bn * 128 + r0) * K + v8off;
  const bf16_t* Bb2 = Bb + (size_t)64 * K;

  const v8f vzero = {0.f, 0.f, 0.f, 0.f, 0.f, 0.f, 0.f, 0.f};
  v8f acc[8];
#pragma unroll
  for (int i = 0; i < 8; ++i) acc[i] = vzero;

  const int nch = K / KC;

#if HAVE_ASYNC_LDS
  async_cp16(Ab,  &sA[0][r0][v8off]);
  async_cp16(Ab2, &sA[0][r0 + 64][v8off]);
  async_cp16(Bb,  &sB[0][r0][v8off]);
  async_cp16(Bb2, &sB[0][r0 + 64][v8off]);
  stage_sync();
  for (int kc = 0; kc < nch; ++kc) {
    const int cur = kc & 1;
    const bool more = (kc + 1) < nch;
    if (more) {
      const int k0 = (kc + 1) * KC;
      async_cp16(Ab + k0,  &sA[cur ^ 1][r0][v8off]);
      async_cp16(Ab2 + k0, &sA[cur ^ 1][r0 + 64][v8off]);
      async_cp16(Bb + k0,  &sB[cur ^ 1][r0][v8off]);
      async_cp16(Bb2 + k0, &sB[cur ^ 1][r0 + 64][v8off]);
    }
    v16bf a = load_a_frag(&sA[cur][w * 16][0], LDS_LD);
#pragma unroll
    for (int nt = 0; nt < 8; ++nt) {
      v16bf b = load_b_frag(&sB[cur][nt * 16][0], LDS_LD);
      acc[nt] = wmma_bf16(a, b, acc[nt]);
    }
    if (more) stage_sync();
  }
#else
  // register-staged double buffer
  uint4 ra0 = *(const uint4*)Ab,  ra1 = *(const uint4*)Ab2;
  uint4 rb0 = *(const uint4*)Bb,  rb1 = *(const uint4*)Bb2;
  *(uint4*)&sA[0][r0][v8off]      = ra0;
  *(uint4*)&sA[0][r0 + 64][v8off] = ra1;
  *(uint4*)&sB[0][r0][v8off]      = rb0;
  *(uint4*)&sB[0][r0 + 64][v8off] = rb1;
  __syncthreads();
  for (int kc = 0; kc < nch; ++kc) {
    const int cur = kc & 1;
    const bool more = (kc + 1) < nch;
    if (more) {
      const int k0 = (kc + 1) * KC;
      ra0 = *(const uint4*)(Ab + k0);  ra1 = *(const uint4*)(Ab2 + k0);
      rb0 = *(const uint4*)(Bb + k0);  rb1 = *(const uint4*)(Bb2 + k0);
    }
    v16bf a = load_a_frag(&sA[cur][w * 16][0], LDS_LD);
#pragma unroll
    for (int nt = 0; nt < 8; ++nt) {
      v16bf b = load_b_frag(&sB[cur][nt * 16][0], LDS_LD);
      acc[nt] = wmma_bf16(a, b, acc[nt]);
    }
    if (more) {
      *(uint4*)&sA[cur ^ 1][r0][v8off]      = ra0;
      *(uint4*)&sA[cur ^ 1][r0 + 64][v8off] = ra1;
      *(uint4*)&sB[cur ^ 1][r0][v8off]      = rb0;
      *(uint4*)&sB[cur ^ 1][r0 + 64][v8off] = rb1;
      __syncthreads();
    }
  }
#endif

  // Epilogue per C/D layout: lane -> n=lane%16, m=(lane/16)*8 + r (vgpr r)
  const int m_base = bm * 128 + w * 16 + (lane >> 4) * 8;
  const int n_sub = lane & 15;
#pragma unroll
  for (int nt = 0; nt < 8; ++nt) {
    int n = bn * 128 + nt * 16 + n_sub;
    float bv = bias ? bias[n] : 0.f;
#pragma unroll
    for (int r = 0; r < 8; ++r)
      C[(size_t)(m_base + r) * N + n] = acc[nt][r] + bv;
  }
}

// ---------------------------------------------------------------------------
// Fused GRU step (one timestep, both directions via blockIdx.y).
// Per block: 32 hidden columns; computes gate tiles G[r|z|n] (64x96, K=1024)
// from the bf16 shadow of h, then fuses sigmoid/tanh/h-update via LDS.
// h and its bf16 shadow are parity double-buffered across step launches.
// ---------------------------------------------------------------------------
__global__ __launch_bounds__(256) void gru_step(
    const float* __restrict__ xp,      // (T,B,3H) f32
    const bf16_t* __restrict__ U,      // (3H,H) bf16 [Ur;Uz;Un]
    const float* __restrict__ bn,      // (H)
    float* __restrict__ hbuf,          // (2 par, 2 dir, B, H) f32
    bf16_t* __restrict__ hbuf_bf,      // (2 par, 2 dir, B, H) bf16
    bf16_t* __restrict__ out_bf,       // (T,B,2H) bf16 or null
    float*  __restrict__ out_f,        // (T,B,2H) f32  or null
    int t)
{
  const int dir = blockIdx.y;
  const int tt = dir ? (T_SEQ - 1 - t) : t;
  const int j0 = blockIdx.x * JB;
  const float* xpt = xp + (size_t)tt * BATCH * 3 * H_DIM;
  const size_t in_off  = (size_t)(((t & 1) * 2) + dir) * BATCH * H_DIM;
  const size_t out_off = (size_t)((((t + 1) & 1) * 2) + dir) * BATCH * H_DIM;
  const float*  hin     = hbuf + in_off;
  float*        hout    = hbuf + out_off;
  const bf16_t* hin_bf  = hbuf_bf + in_off;
  bf16_t*       hout_bf = hbuf_bf + out_off;

  __shared__ __align__(16) __bf16 sH[2][BATCH][LDS_LD];    // 2 x 64 x 40
  __shared__ __align__(16) __bf16 sU[2][3 * JB][LDS_LD];   // 2 x 96 x 40
  __shared__ __align__(16) float  sG[BATCH][3 * JB + 4];   // 64 x 100 f32

  const int tid = threadIdx.x;
  const int w = tid >> 5, lane = tid & 31;
  const int m_blk = w & 3;   // 16-row block of the 64 batch rows
  const int nst   = w >> 2;  // 0/1 -> column tiles {0,1,2} / {3,4,5}

  // staging addresses: sH = 256 16B vectors (1/thread);
  // sU = 384 16B vectors (idx = tid, and tid+256 for tid<128)
  const int hr = tid >> 2, hv = (tid & 3) * 8;
  const bf16_t* hsrc = hin_bf + (size_t)hr * H_DIM + hv;
  const int ur0 = tid >> 2;                 // U rows 0..63
  const int g0u = ur0 / JB, jj0u = ur0 - g0u * JB;
  const bf16_t* usrc0 = U + (size_t)(g0u * H_DIM + j0 + jj0u) * H_DIM + hv;
  const int ur1 = 64 + (tid >> 2);          // U rows 64..95 (tid<128)
  const int jj1u = ur1 - 64;                // gate 2
  const bf16_t* usrc1 = U + (size_t)(2 * H_DIM + j0 + jj1u) * H_DIM + hv;

  const v8f vzero = {0.f, 0.f, 0.f, 0.f, 0.f, 0.f, 0.f, 0.f};
  v8f acc[3];
#pragma unroll
  for (int i = 0; i < 3; ++i) acc[i] = vzero;

  const int nch = H_DIM / KC;

#if HAVE_ASYNC_LDS
  async_cp16(hsrc,  &sH[0][hr][hv]);
  async_cp16(usrc0, &sU[0][ur0][hv]);
  if (tid < 128) async_cp16(usrc1, &sU[0][ur1][hv]);
  stage_sync();
  for (int kc = 0; kc < nch; ++kc) {
    const int cur = kc & 1;
    const bool more = (kc + 1) < nch;
    if (more) {
      const int k0 = (kc + 1) * KC;
      async_cp16(hsrc + k0,  &sH[cur ^ 1][hr][hv]);
      async_cp16(usrc0 + k0, &sU[cur ^ 1][ur0][hv]);
      if (tid < 128) async_cp16(usrc1 + k0, &sU[cur ^ 1][ur1][hv]);
    }
    v16bf a = load_a_frag(&sH[cur][m_blk * 16][0], LDS_LD);
#pragma unroll
    for (int ct = 0; ct < 3; ++ct) {
      v16bf b = load_b_frag(&sU[cur][(nst * 3 + ct) * 16][0], LDS_LD);
      acc[ct] = wmma_bf16(a, b, acc[ct]);
    }
    if (more) stage_sync();
  }
#else
  uint4 rh = *(const uint4*)hsrc;
  uint4 ru0 = *(const uint4*)usrc0;
  uint4 ru1 = {0, 0, 0, 0};
  if (tid < 128) ru1 = *(const uint4*)usrc1;
  *(uint4*)&sH[0][hr][hv] = rh;
  *(uint4*)&sU[0][ur0][hv] = ru0;
  if (tid < 128) *(uint4*)&sU[0][ur1][hv] = ru1;
  __syncthreads();
  for (int kc = 0; kc < nch; ++kc) {
    const int cur = kc & 1;
    const bool more = (kc + 1) < nch;
    if (more) {
      const int k0 = (kc + 1) * KC;
      rh = *(const uint4*)(hsrc + k0);
      ru0 = *(const uint4*)(usrc0 + k0);
      if (tid < 128) ru1 = *(const uint4*)(usrc1 + k0);
    }
    v16bf a = load_a_frag(&sH[cur][m_blk * 16][0], LDS_LD);
#pragma unroll
    for (int ct = 0; ct < 3; ++ct) {
      v16bf b = load_b_frag(&sU[cur][(nst * 3 + ct) * 16][0], LDS_LD);
      acc[ct] = wmma_bf16(a, b, acc[ct]);
    }
    if (more) {
      *(uint4*)&sH[cur ^ 1][hr][hv] = rh;
      *(uint4*)&sU[cur ^ 1][ur0][hv] = ru0;
      if (tid < 128) *(uint4*)&sU[cur ^ 1][ur1][hv] = ru1;
      __syncthreads();
    }
  }
#endif

  // dump gate accumulators so each thread can gather its (r,z,n) triple
  {
    const int m_base = m_blk * 16 + (lane >> 4) * 8;
    const int n_sub = lane & 15;
#pragma unroll
    for (int ct = 0; ct < 3; ++ct) {
      int n = (nst * 3 + ct) * 16 + n_sub;
#pragma unroll
      for (int r = 0; r < 8; ++r) sG[m_base + r][n] = acc[ct][r];
    }
  }
  __syncthreads();

  // fused GRU cell elementwise: 64 x 32 elements, 8 per thread
  for (int idx = tid; idx < BATCH * JB; idx += 256) {
    int b = idx / JB, jj = idx - b * JB;
    int j = j0 + jj;
    float gr = sG[b][jj];
    float gz = sG[b][JB + jj];
    float gn = sG[b][2 * JB + jj];
    const float* xr = xpt + (size_t)b * 3 * H_DIM;
    float rx = xr[j];
    float zx = xr[H_DIM + j];
    float nx = xr[2 * H_DIM + j];
    float hold = hin[(size_t)b * H_DIM + j];
    float r = 1.f / (1.f + __expf(-(rx + gr)));
    float z = 1.f / (1.f + __expf(-(zx + gz)));
    float n = tanhf(nx + r * (gn + bn[j]));
    float hn = (1.f - z) * n + z * hold;
    hout[(size_t)b * H_DIM + j] = hn;
    hout_bf[(size_t)b * H_DIM + j] = (__bf16)hn;
    size_t oo = (size_t)tt * BATCH * 2 * H_DIM + (size_t)b * 2 * H_DIM +
                (size_t)dir * H_DIM + j;
    if (out_bf) out_bf[oo] = (__bf16)hn;
    if (out_f)  out_f[oo]  = hn;
  }
}

// ---------------------------------------------------------------------------
// utility kernels
// ---------------------------------------------------------------------------
__global__ void f32_to_bf16_kernel(const float* __restrict__ src,
                                   bf16_t* __restrict__ dst, size_t n) {
  size_t i = (size_t)blockIdx.x * blockDim.x + threadIdx.x;
  size_t stride = (size_t)gridDim.x * blockDim.x;
  for (; i < n; i += stride) dst[i] = (__bf16)src[i];
}

__global__ void zero_u32_kernel(unsigned* __restrict__ p, size_t nwords) {
  size_t i = (size_t)blockIdx.x * blockDim.x + threadIdx.x;
  size_t stride = (size_t)gridDim.x * blockDim.x;
  for (; i < nwords; i += stride) p[i] = 0u;
}

// ---------------------------------------------------------------------------
// host launcher
// ---------------------------------------------------------------------------
extern "C" void kernel_launch(void* const* d_in, const int* in_sizes, int n_in,
                              void* d_out, int out_size, void* d_ws, size_t ws_size,
                              hipStream_t stream) {
  (void)in_sizes; (void)n_in; (void)out_size; (void)ws_size;
  const int T = T_SEQ, B = BATCH, IN = IN_DIM, H = H_DIM;
  const size_t M = (size_t)T * B;  // 16384 GEMM rows

  const float* x   = (const float*)d_in[0];
  const float* W0  = (const float*)d_in[1];
  const float* b0  = (const float*)d_in[2];
  const float* Ur0 = (const float*)d_in[3];
  const float* Uz0 = (const float*)d_in[4];
  const float* Un0 = (const float*)d_in[5];
  const float* bn0 = (const float*)d_in[6];
  const float* W1  = (const float*)d_in[7];
  const float* b1  = (const float*)d_in[8];
  const float* Ur1 = (const float*)d_in[9];
  const float* Uz1 = (const float*)d_in[10];
  const float* Un1 = (const float*)d_in[11];
  const float* bn1 = (const float*)d_in[12];
  float* out = (float*)d_out;

  char* ws = (char*)d_ws;
  size_t off = 0;
  auto carve = [&](size_t bytes) -> void* {
    void* p = ws + off;
    off += (bytes + 255) & ~(size_t)255;
    return p;
  };
  bf16_t* x_bf    = (bf16_t*)carve(M * IN * 2);                 // 33.5 MB
  bf16_t* W0_bf   = (bf16_t*)carve((size_t)3 * H * IN * 2);     //  6.3 MB
  bf16_t* W1_bf   = (bf16_t*)carve((size_t)3 * H * 2 * H * 2);  // 12.6 MB
  bf16_t* U0_bf   = (bf16_t*)carve((size_t)3 * H * H * 2);      //  6.3 MB
  bf16_t* U1_bf   = (bf16_t*)carve((size_t)3 * H * H * 2);      //  6.3 MB
  bf16_t* out0_bf = (bf16_t*)carve(M * 2 * H * 2);              // 67.1 MB
  float*  xp      = (float*)carve(M * 3 * H * 4);               // 201.3 MB (reused)
  float*  h0      = (float*)carve((size_t)2 * 2 * B * H * 4);   // (par,dir,B,H)
  float*  h1      = (float*)carve((size_t)2 * 2 * B * H * 4);
  bf16_t* h0_bf   = (bf16_t*)carve((size_t)2 * 2 * B * H * 2);
  bf16_t* h1_bf   = (bf16_t*)carve((size_t)2 * 2 * B * H * 2);

  const dim3 cgrid(2048), cblk(256);
  f32_to_bf16_kernel<<<cgrid, cblk, 0, stream>>>(x, x_bf, M * IN);
  f32_to_bf16_kernel<<<cgrid, cblk, 0, stream>>>(W0, W0_bf, (size_t)3 * H * IN);
  f32_to_bf16_kernel<<<cgrid, cblk, 0, stream>>>(W1, W1_bf, (size_t)3 * H * 2 * H);
  f32_to_bf16_kernel<<<cgrid, cblk, 0, stream>>>(Ur0, U0_bf,                     (size_t)H * H);
  f32_to_bf16_kernel<<<cgrid, cblk, 0, stream>>>(Uz0, U0_bf + (size_t)H * H,     (size_t)H * H);
  f32_to_bf16_kernel<<<cgrid, cblk, 0, stream>>>(Un0, U0_bf + (size_t)2 * H * H, (size_t)H * H);
  f32_to_bf16_kernel<<<cgrid, cblk, 0, stream>>>(Ur1, U1_bf,                     (size_t)H * H);
  f32_to_bf16_kernel<<<cgrid, cblk, 0, stream>>>(Uz1, U1_bf + (size_t)H * H,     (size_t)H * H);
  f32_to_bf16_kernel<<<cgrid, cblk, 0, stream>>>(Un1, U1_bf + (size_t)2 * H * H, (size_t)H * H);
  // zero initial h (parity-0 blocks, f32 + bf16 shadows)
  zero_u32_kernel<<<dim3(512), cblk, 0, stream>>>((unsigned*)h0,    (size_t)2 * B * H);
  zero_u32_kernel<<<dim3(512), cblk, 0, stream>>>((unsigned*)h1,    (size_t)2 * B * H);
  zero_u32_kernel<<<dim3(512), cblk, 0, stream>>>((unsigned*)h0_bf, (size_t)B * H);
  zero_u32_kernel<<<dim3(512), cblk, 0, stream>>>((unsigned*)h1_bf, (size_t)B * H);

  // xp0 = x @ W0^T + b0 : M=16384, N=3072, K=1024
  {
    dim3 grid(3 * H / 128, (int)(M / 128));
    gemm_bf16_bias<<<grid, 256, 0, stream>>>(x_bf, W0_bf, b0, xp,
                                             (int)M, 3 * H, IN);
  }
  // layer-0 bidirectional scan (writes out0 in bf16)
  {
    dim3 grid(H / JB, 2);
    for (int t = 0; t < T; ++t)
      gru_step<<<grid, 256, 0, stream>>>(xp, U0_bf, bn0, h0, h0_bf,
                                         out0_bf, (float*)nullptr, t);
  }
  // xp1 = out0 @ W1^T + b1 : M=16384, N=3072, K=2048 (reuses xp buffer)
  {
    dim3 grid(3 * H / 128, (int)(M / 128));
    gemm_bf16_bias<<<grid, 256, 0, stream>>>(out0_bf, W1_bf, b1, xp,
                                             (int)M, 3 * H, 2 * H);
  }
  // layer-1 bidirectional scan (writes out1 f32 directly into d_out)
  {
    dim3 grid(H / JB, 2);
    for (int t = 0; t < T; ++t)
      gru_step<<<grid, 256, 0, stream>>>(xp, U1_bf, bn1, h1, h1_bf,
                                         (bf16_t*)nullptr, out, t);
  }

  // hidden = [hf0, hb0, hf1, hb1] : final states live in parity-0 blocks
  float* hid = out + M * 2 * H;
  (void)hipMemcpyAsync(hid, h0, (size_t)2 * B * H * sizeof(float),
                       hipMemcpyDeviceToDevice, stream);
  (void)hipMemcpyAsync(hid + (size_t)2 * B * H, h1,
                       (size_t)2 * B * H * sizeof(float),
                       hipMemcpyDeviceToDevice, stream);
}
